// RFCN_17119739642285
// MI455X (gfx1250) — compile-verified, matrix operand order checked by reading, not run on hardware
//
#include <hip/hip_runtime.h>
#include <hip/hip_bf16.h>
#include <math.h>

// ---------------- problem constants (fixed by setup_inputs) ----------------
#define NUM_CLASSES 21
#define POST_NMS    1000
#define MAX_DETS    100
#define NEGV        (-1000000000.0f)
#define SCORE_TH    0.05f
#define FINAL_NMS_TH 0.5f
#define RPN_NMS_TH  0.7f

#define BATCH   2
#define NSP     1024          // 32*32 spatial positions
#define PADW    34            // padded NHWC width/height
#define CFEAT   2048
#define CMID    512
#define CCLS    189           // 21*9
#define CCLS_P  192           // padded to 16
#define KBB     768           // 3*16*16
#define KRPN    18432         // 2048*9
#define ANCH_TOT 9216         // 1024*9
#define NROIS   2000          // BATCH*POST_NMS

typedef __attribute__((ext_vector_type(16))) __bf16       v16bf;
typedef __attribute__((ext_vector_type(8)))  float        v8f;
typedef __attribute__((ext_vector_type(4)))  unsigned int v4u;

// ---------------- helpers ----------------
__device__ __forceinline__ unsigned short f2bf(float f) {
    unsigned int u = __float_as_uint(f);
    unsigned int lsb = (u >> 16) & 1u;
    u += 0x7fffu + lsb;                 // round-to-nearest-even
    return (unsigned short)(u >> 16);
}

__device__ __forceinline__ v16bf load_frag16(const unsigned short* p, int khalf) {
    // 16-bit WMMA operand layout: per lane two contiguous 8-half chunks (2x b128)
    union { v4u u[2]; v16bf v; } f;
    f.u[0] = *reinterpret_cast<const v4u*>(p + 8 * khalf);
    f.u[1] = *reinterpret_cast<const v4u*>(p + 16 + 8 * khalf);
    return f.v;
}

__device__ __forceinline__ float iou4(float bx1, float by1, float bx2, float by2,
                                      float ox1, float oy1, float ox2, float oy2) {
    float x1 = fmaxf(bx1, ox1), y1 = fmaxf(by1, oy1);
    float x2 = fminf(bx2, ox2), y2 = fminf(by2, oy2);
    float inter = fmaxf(x2 - x1, 0.f) * fmaxf(y2 - y1, 0.f);
    float a1 = (bx2 - bx1) * (by2 - by1);
    float a2 = (ox2 - ox1) * (oy2 - oy1);
    return inter / (a1 + a2 - inter + 1e-9f);
}

// wave32 butterfly argmax (no barriers)
__device__ __forceinline__ void argmax_wave(float& v, int& i) {
#pragma unroll
    for (int o = 16; o > 0; o >>= 1) {
        float ov = __shfl_xor(v, o, 32);
        int   oi = __shfl_xor(i, o, 32);
        if (ov > v) { v = ov; i = oi; }
    }
}

// ---------------- packing kernels ----------------
__global__ void cvt_bf16_k(const float* __restrict__ src, unsigned short* __restrict__ dst, int n) {
    int i = blockIdx.x * blockDim.x + threadIdx.x;
    if (i < n) dst[i] = f2bf(src[i]);
}

// w_rpn [512][2048][3][3] f32 -> [512][kk=ky*3+kx][ci] bf16  (K contiguous per out row)
__global__ void repack_wrpn_k(const float* __restrict__ src, unsigned short* __restrict__ dst, int n) {
    int t = blockIdx.x * blockDim.x + threadIdx.x;
    if (t >= n) return;
    int o  = t / KRPN;
    int k  = t - o * KRPN;
    int kk = k >> 11;          // /2048
    int ci = k & 2047;
    dst[t] = f2bf(src[(o * 2048 + ci) * 9 + kk]);
}

// images (2,3,512,512) -> patches [b][n=(y,x)][k=(c,ky,kx)] bf16 (stride-16 16x16 VALID = reshape)
__global__ void im2col_bb_k(const float* __restrict__ img, unsigned short* __restrict__ dst, int n) {
    int t = blockIdx.x * blockDim.x + threadIdx.x;
    if (t >= n) return;
    int b   = t / (NSP * KBB);
    int r   = t - b * (NSP * KBB);
    int nsp = r / KBB;
    int k   = r - nsp * KBB;
    int y = nsp >> 5, x = nsp & 31;
    int c  = k >> 8;
    int kk = k & 255;
    int ky = kk >> 4, kx = kk & 15;
    dst[t] = f2bf(img[((b * 3 + c) * 512 + (y * 16 + ky)) * 512 + (x * 16 + kx)]);
}

// ---------------- WMMA GEMM core ----------------
// Inner pipelined K-loop: L multiple of 64; double-buffered fragment sets so the
// next chunk's global_load_b128s are in flight while the current chunk's WMMAs run.
__device__ __forceinline__ void gemm_inner(v8f acc[4], const unsigned short* a,
                                           const unsigned short* b0, const unsigned short* b1,
                                           const unsigned short* b2, const unsigned short* b3,
                                           int L, int khalf) {
    const unsigned short* br[4] = {b0, b1, b2, b3};
    v16bf A0 = load_frag16(a, khalf);
    v16bf B0[4];
#pragma unroll
    for (int t = 0; t < 4; ++t) B0[t] = load_frag16(br[t], khalf);

    for (int k = 0; k < L; k += 64) {
        v16bf A1 = load_frag16(a + k + 32, khalf);
        v16bf B1[4];
#pragma unroll
        for (int t = 0; t < 4; ++t) B1[t] = load_frag16(br[t] + k + 32, khalf);
#pragma unroll
        for (int t = 0; t < 4; ++t)
            acc[t] = __builtin_amdgcn_wmma_f32_16x16x32_bf16(
                false, A0, false, B0[t], (short)0, acc[t], false, false);
        if (k + 64 < L) {
            A0 = load_frag16(a + k + 64, khalf);
#pragma unroll
            for (int t = 0; t < 4; ++t) B0[t] = load_frag16(br[t] + k + 64, khalf);
        }
#pragma unroll
        for (int t = 0; t < 4; ++t)
            acc[t] = __builtin_amdgcn_wmma_f32_16x16x32_bf16(
                false, A1, false, B1[t], (short)0, acc[t], false, false);
    }
}

// C[Mpad,1024] = A[Mpad,K] * B + bias.  A and C are padded to Mpad (multiple of 16):
// no guards anywhere.  One wave computes a 16x64 C tile (A fragment reused 4x).
// BMODE: 0 = B is [N][K] row-major; 1 = padded NHWC feats, 3x3 conv (K=9*2048);
//        2 = padded NHWC feats, 1x1 conv (K=2048)
// SMODE: 0 = f32 [Mpad][1024]; 1 = bf16 padded NHWC (Mpad=channels); 2 = bf16 NHWC [1024][Mpad]
template <int BMODE, int SMODE, int RELU>
__global__ __launch_bounds__(128)
void gemm_wmma_k(const unsigned short* __restrict__ A,
                 const unsigned short* __restrict__ B,
                 const float* __restrict__ bias,
                 void* __restrict__ Cout,
                 int Mpad, int Kdim,
                 long long bStride, long long cStride) {
    int lane  = threadIdx.x;          // 0..31
    int wave  = threadIdx.y;          // 0..3
    int l15   = lane & 15;
    int khalf = lane >> 4;
    int n0 = (blockIdx.x * 4 + wave) * 64;
    int m0 = blockIdx.y * 16;
    int b  = blockIdx.z;

    const unsigned short* Bb = B + (long long)b * bStride;

    v8f acc[4];
    v8f z8 = {0.f, 0.f, 0.f, 0.f, 0.f, 0.f, 0.f, 0.f};
#pragma unroll
    for (int t = 0; t < 4; ++t) acc[t] = z8;

    const unsigned short* arow = A + (long long)(m0 + l15) * Kdim;

    int colY[4], colX[4];
#pragma unroll
    for (int t = 0; t < 4; ++t) {
        int col = n0 + t * 16 + l15;
        colY[t] = col >> 5; colX[t] = col & 31;
    }

    if constexpr (BMODE == 1) {
        for (int kk = 0; kk < 9; ++kk) {
            int ky = kk / 3, kx = kk - (kk / 3) * 3;
            const unsigned short* br[4];
#pragma unroll
            for (int t = 0; t < 4; ++t)
                br[t] = Bb + (long long)((colY[t] + ky) * PADW + (colX[t] + kx)) * 2048;
            gemm_inner(acc, arow + kk * 2048, br[0], br[1], br[2], br[3], 2048, khalf);
        }
    } else if constexpr (BMODE == 2) {
        const unsigned short* br[4];
#pragma unroll
        for (int t = 0; t < 4; ++t)
            br[t] = Bb + (long long)((colY[t] + 1) * PADW + (colX[t] + 1)) * 2048;
        gemm_inner(acc, arow, br[0], br[1], br[2], br[3], Kdim, khalf);
    } else {
        const unsigned short* br[4];
#pragma unroll
        for (int t = 0; t < 4; ++t)
            br[t] = Bb + (long long)(n0 + t * 16 + l15) * Kdim;
        gemm_inner(acc, arow, br[0], br[1], br[2], br[3], Kdim, khalf);
    }

#pragma unroll
    for (int t = 0; t < 4; ++t) {
        int col = n0 + t * 16 + l15;
#pragma unroll
        for (int j = 0; j < 8; ++j) {
            int row = m0 + j + 8 * khalf;
            float v = acc[t][j] + bias[row];
            if (RELU) v = fmaxf(v, 0.f);
            if constexpr (SMODE == 0) {
                ((float*)Cout)[(long long)b * cStride + (long long)row * NSP + col] = v;
            } else if constexpr (SMODE == 1) {
                int y = col >> 5, x = col & 31;
                ((unsigned short*)Cout)[(long long)b * cStride +
                    (long long)((y + 1) * PADW + (x + 1)) * Mpad + row] = f2bf(v);
            } else {
                ((unsigned short*)Cout)[(long long)b * cStride + (long long)col * Mpad + row] = f2bf(v);
            }
        }
    }
}

// ---------------- anchors + decode + clip + minsize ----------------
__global__ void decode_k(const float* __restrict__ Cobj, const float* __restrict__ Cbox,
                         float* __restrict__ boxes, float* __restrict__ scores, int n) {
    int t = blockIdx.x * blockDim.x + threadIdx.x;
    if (t >= n) return;
    int b = t / ANCH_TOT;
    int r = t - b * ANCH_TOT;
    int nsp = r / 9;
    int a   = r - nsp * 9;
    int y = nsp >> 5, x = nsp & 31;
    const float scl[3] = {8.f, 16.f, 32.f};
    const float rrs[3] = {0.70710678f, 1.0f, 1.41421356f};
    int ri = a / 3, si = a - ri * 3;
    float wa  = 16.f * scl[si] / rrs[ri];
    float ha  = 16.f * scl[si] * rrs[ri];
    float cxa = (x + 0.5f) * 16.f;
    float cya = (y + 0.5f) * 16.f;

    float obj = Cobj[(b * 16 + a) * NSP + nsp];               // Cobj padded to 16 rows
    float d0 = Cbox[(b * 48 + a * 4 + 0) * NSP + nsp];        // Cbox padded to 48 rows
    float d1 = Cbox[(b * 48 + a * 4 + 1) * NSP + nsp];
    float d2 = Cbox[(b * 48 + a * 4 + 2) * NSP + nsp];
    float d3 = Cbox[(b * 48 + a * 4 + 3) * NSP + nsp];

    float cx = d0 * wa + cxa;
    float cy = d1 * ha + cya;
    float w  = wa * expf(fminf(fmaxf(d2, -4.f), 4.f));
    float h  = ha * expf(fminf(fmaxf(d3, -4.f), 4.f));
    float x1 = fminf(fmaxf(cx - 0.5f * w, 0.f), 511.f);
    float y1 = fminf(fmaxf(cy - 0.5f * h, 0.f), 511.f);
    float x2 = fminf(fmaxf(cx + 0.5f * w, 0.f), 511.f);
    float y2 = fminf(fmaxf(cy + 0.5f * h, 0.f), 511.f);
    float wb = x2 - x1, hb = y2 - y1;
    scores[t] = (wb >= 16.f && hb >= 16.f) ? obj : NEGV;
    boxes[t * 4 + 0] = x1; boxes[t * 4 + 1] = y1;
    boxes[t * 4 + 2] = x2; boxes[t * 4 + 3] = y2;
}

// ---------------- RPN NMS: 1 workgroup per batch, scores in LDS ----------------
__global__ __launch_bounds__(1024)
void rpn_nms_k(const float* __restrict__ scoresAll, const float* __restrict__ boxesAll,
               float* __restrict__ roi_boxes) {
    __shared__ float ss[ANCH_TOT];
    __shared__ float rv[32];
    __shared__ int   ri[32];
    __shared__ float bbx[4];
    int b   = blockIdx.x;
    int tid = threadIdx.x;
    int wid = tid >> 5, lid = tid & 31;
    const float* sc = scoresAll + b * ANCH_TOT;
    const float* bx = boxesAll + (long long)b * ANCH_TOT * 4;
    for (int i = tid; i < ANCH_TOT; i += 1024) ss[i] = sc[i];
    __syncthreads();
    for (int it = 0; it < POST_NMS; ++it) {
        float lm = -3.0e9f; int li = 0;
        for (int i = tid; i < ANCH_TOT; i += 1024) {
            float v = ss[i];
            if (v > lm) { lm = v; li = i; }
        }
        argmax_wave(lm, li);
        if (lid == 0) { rv[wid] = lm; ri[wid] = li; }
        __syncthreads();
        if (wid == 0) {
            float v = rv[lid]; int i2 = ri[lid];    // 32 waves -> 32 entries
            argmax_wave(v, i2);
            if (lid == 0) {
                float b0 = bx[i2 * 4 + 0], b1 = bx[i2 * 4 + 1];
                float b2 = bx[i2 * 4 + 2], b3 = bx[i2 * 4 + 3];
                bbx[0] = b0; bbx[1] = b1; bbx[2] = b2; bbx[3] = b3;
                ss[i2] = NEGV;
                bool valid = v > NEGV * 0.5f;
                float* out = roi_boxes + (b * POST_NMS + it) * 4;
                out[0] = valid ? b0 : 0.f; out[1] = valid ? b1 : 0.f;
                out[2] = valid ? b2 : 0.f; out[3] = valid ? b3 : 0.f;
            }
        }
        __syncthreads();
        float b0 = bbx[0], b1 = bbx[1], b2 = bbx[2], b3 = bbx[3];
        for (int i = tid; i < ANCH_TOT; i += 1024) {
            float o0 = bx[i * 4 + 0], o1 = bx[i * 4 + 1];
            float o2 = bx[i * 4 + 2], o3 = bx[i * 4 + 3];
            if (iou4(b0, b1, b2, b3, o0, o1, o2, o3) > RPN_NMS_TH) ss[i] = NEGV;
        }
        __syncthreads();
    }
}

// ---------------- PSROI pooling ----------------
__global__ void psroi_k(const float* __restrict__ Ccls, const float* __restrict__ roi_boxes,
                        float* __restrict__ outS) {
    int t = blockIdx.x * blockDim.x + threadIdx.x;
    if (t >= NROIS * NUM_CLASSES) return;
    int r = t / NUM_CLASSES, c = t - r * NUM_CLASSES;
    int b = r / POST_NMS;
    const float* bp = roi_boxes + r * 4;
    float x1 = bp[0] * 0.0625f, y1 = bp[1] * 0.0625f;
    float x2 = bp[2] * 0.0625f, y2 = bp[3] * 0.0625f;
    float bw = fmaxf(x2 - x1, 0.001f) * (1.f / 3.f);
    float bh = fmaxf(y2 - y1, 0.001f) * (1.f / 3.f);
    float sum = 0.f;
    for (int i = 0; i < 3; ++i) {
        for (int j = 0; j < 3; ++j) {
            float px = fminf(fmaxf(x1 + bw * (j + 0.5f), 0.f), 31.f);
            float py = fminf(fmaxf(y1 + bh * (i + 0.5f), 0.f), 31.f);
            int x0 = (int)floorf(px), y0 = (int)floorf(py);
            int x1i = min(x0 + 1, 31), y1i = min(y0 + 1, 31);
            float wx = px - (float)x0, wy = py - (float)y0;
            const float* m = Ccls + ((long long)b * CCLS_P + (c * 9 + i * 3 + j)) * NSP;
            sum += m[y0 * 32 + x0]  * (1.f - wy) * (1.f - wx)
                 + m[y0 * 32 + x1i] * (1.f - wy) * wx
                 + m[y1i * 32 + x0] * wy * (1.f - wx)
                 + m[y1i * 32 + x1i] * wy * wx;
        }
    }
    outS[t] = sum * (1.f / 9.f);
}

// ---------------- softmax over 21 classes ----------------
__global__ void softmax_k(const float* __restrict__ s, float* __restrict__ p) {
    int r = blockIdx.x * blockDim.x + threadIdx.x;
    if (r >= NROIS) return;
    float mx = -3.0e38f;
    for (int c = 0; c < NUM_CLASSES; ++c) mx = fmaxf(mx, s[r * NUM_CLASSES + c]);
    float e[NUM_CLASSES]; float sum = 0.f;
    for (int c = 0; c < NUM_CLASSES; ++c) { e[c] = expf(s[r * NUM_CLASSES + c] - mx); sum += e[c]; }
    float inv = 1.f / sum;
    for (int c = 0; c < NUM_CLASSES; ++c) p[r * NUM_CLASSES + c] = e[c] * inv;
}

// ---------------- per-class NMS: 1 workgroup (8 waves) per class ----------------
__global__ __launch_bounds__(256)
void class_nms_k(const float* __restrict__ probs, const float* __restrict__ roi_boxes,
                 float* __restrict__ clsS, float* __restrict__ clsB) {
    __shared__ float ss[NROIS];
    __shared__ float rv[8];
    __shared__ int   ri[8];
    __shared__ float bbx[4];
    int cb  = blockIdx.x;                  // class label = cb + 1
    int tid = threadIdx.x;
    int wid = tid >> 5, lid = tid & 31;
    for (int i = tid; i < NROIS; i += 256) {
        float v = probs[i * NUM_CLASSES + cb + 1];
        ss[i] = (v > SCORE_TH) ? v : NEGV;
    }
    __syncthreads();
    for (int it = 0; it < MAX_DETS; ++it) {
        float lm = -3.0e9f; int li = 0;
        for (int i = tid; i < NROIS; i += 256) {
            float v = ss[i];
            if (v > lm) { lm = v; li = i; }
        }
        argmax_wave(lm, li);
        if (lid == 0) { rv[wid] = lm; ri[wid] = li; }
        __syncthreads();
        if (wid == 0) {
            float v = (lid < 8) ? rv[lid] : -3.0e9f;
            int i2  = (lid < 8) ? ri[lid] : 0;
            argmax_wave(v, i2);
            if (lid == 0) {
                float b0 = roi_boxes[i2 * 4 + 0], b1 = roi_boxes[i2 * 4 + 1];
                float b2 = roi_boxes[i2 * 4 + 2], b3 = roi_boxes[i2 * 4 + 3];
                bbx[0] = b0; bbx[1] = b1; bbx[2] = b2; bbx[3] = b3;
                ss[i2] = NEGV;
                int o = cb * MAX_DETS + it;
                clsS[o] = v;
                clsB[o * 4 + 0] = b0; clsB[o * 4 + 1] = b1;
                clsB[o * 4 + 2] = b2; clsB[o * 4 + 3] = b3;
            }
        }
        __syncthreads();
        float b0 = bbx[0], b1 = bbx[1], b2 = bbx[2], b3 = bbx[3];
        for (int i = tid; i < NROIS; i += 256) {
            float o0 = roi_boxes[i * 4 + 0], o1 = roi_boxes[i * 4 + 1];
            float o2 = roi_boxes[i * 4 + 2], o3 = roi_boxes[i * 4 + 3];
            if (iou4(b0, b1, b2, b3, o0, o1, o2, o3) > FINAL_NMS_TH) ss[i] = NEGV;
        }
        __syncthreads();
    }
}

// ---------------- final top-100 over 20*100 detections ----------------
__global__ __launch_bounds__(256)
void final_topk_k(const float* __restrict__ clsS, const float* __restrict__ clsB,
                  float* __restrict__ out) {
    __shared__ float ss[NROIS];          // 2000 = 20*100
    __shared__ float rv[8];
    __shared__ int   ri[8];
    int tid = threadIdx.x;
    int wid = tid >> 5, lid = tid & 31;
    for (int i = tid; i < 20 * MAX_DETS; i += 256) ss[i] = clsS[i];
    __syncthreads();
    for (int it = 0; it < MAX_DETS; ++it) {
        float lm = -3.0e9f; int li = 0;
        for (int i = tid; i < 20 * MAX_DETS; i += 256) {
            float v = ss[i];
            if (v > lm) { lm = v; li = i; }
        }
        argmax_wave(lm, li);
        if (lid == 0) { rv[wid] = lm; ri[wid] = li; }
        __syncthreads();
        if (wid == 0) {
            float v = (lid < 8) ? rv[lid] : -3.0e9f;
            int i2  = (lid < 8) ? ri[lid] : 0;
            argmax_wave(v, i2);
            if (lid == 0) {
                bool valid = v > SCORE_TH;
                out[it * 4 + 0] = valid ? clsB[i2 * 4 + 0] : 0.f;
                out[it * 4 + 1] = valid ? clsB[i2 * 4 + 1] : 0.f;
                out[it * 4 + 2] = valid ? clsB[i2 * 4 + 2] : 0.f;
                out[it * 4 + 3] = valid ? clsB[i2 * 4 + 3] : 0.f;
                out[400 + it] = valid ? v : 0.f;
                out[500 + it] = valid ? (float)(i2 / MAX_DETS + 1) : 0.f;
                ss[i2] = NEGV;
            }
        }
        __syncthreads();
    }
}

// ---------------- host launch ----------------
extern "C" void kernel_launch(void* const* d_in, const int* in_sizes, int n_in,
                              void* d_out, int out_size, void* d_ws, size_t ws_size,
                              hipStream_t stream) {
    const float* images = (const float*)d_in[0];
    const float* w_bb   = (const float*)d_in[1];
    const float* b_bb   = (const float*)d_in[2];
    const float* w_rpn  = (const float*)d_in[3];
    const float* b_rpn  = (const float*)d_in[4];
    const float* w_obj  = (const float*)d_in[5];
    const float* b_obj  = (const float*)d_in[6];
    const float* w_box  = (const float*)d_in[7];
    const float* b_box  = (const float*)d_in[8];
    const float* w_cls  = (const float*)d_in[9];
    const float* b_cls  = (const float*)d_in[10];

    char* ws = (char*)d_ws;
    size_t off = 0;
    auto carve = [&](size_t bytes) -> char* {
        char* p = ws + off;
        off = (off + bytes + 255) & ~(size_t)255;
        return p;
    };
    unsigned short* wbbB  = (unsigned short*)carve((size_t)2048 * KBB * 2);
    unsigned short* wrpnB = (unsigned short*)carve((size_t)512 * KRPN * 2);
    unsigned short* wobjB = (unsigned short*)carve((size_t)16 * 512 * 2);      // padded 9->16
    unsigned short* wboxB = (unsigned short*)carve((size_t)48 * 512 * 2);      // padded 36->48
    unsigned short* wclsB = (unsigned short*)carve((size_t)CCLS_P * 2048 * 2); // padded 189->192
    unsigned short* patch = (unsigned short*)carve((size_t)BATCH * NSP * KBB * 2);
    unsigned short* featsP= (unsigned short*)carve((size_t)BATCH * PADW * PADW * CFEAT * 2);
    unsigned short* midT  = (unsigned short*)carve((size_t)BATCH * NSP * CMID * 2);
    float* Cobj   = (float*)carve((size_t)BATCH * 16 * NSP * 4);
    float* Cbox   = (float*)carve((size_t)BATCH * 48 * NSP * 4);
    float* Ccls   = (float*)carve((size_t)BATCH * CCLS_P * NSP * 4);
    float* boxes  = (float*)carve((size_t)BATCH * ANCH_TOT * 4 * 4);
    float* scoresA= (float*)carve((size_t)BATCH * ANCH_TOT * 4);
    float* roiB   = (float*)carve((size_t)NROIS * 4 * 4);
    float* psS    = (float*)carve((size_t)NROIS * NUM_CLASSES * 4);
    float* probs  = (float*)carve((size_t)NROIS * NUM_CLASSES * 4);
    float* clsS   = (float*)carve((size_t)20 * MAX_DETS * 4);
    float* clsB   = (float*)carve((size_t)20 * MAX_DETS * 4 * 4);

    // zero padded weight tails + padded feats border
    hipMemsetAsync(wobjB, 0, (size_t)16 * 512 * 2, stream);
    hipMemsetAsync(wboxB, 0, (size_t)48 * 512 * 2, stream);
    hipMemsetAsync(wclsB, 0, (size_t)CCLS_P * 2048 * 2, stream);
    hipMemsetAsync(featsP, 0, (size_t)BATCH * PADW * PADW * CFEAT * 2, stream);

    // 1) weight conversions / repacks
    { int n = 2048 * KBB;  cvt_bf16_k<<<(n + 255) / 256, 256, 0, stream>>>(w_bb,  wbbB, n); }
    { int n = 9 * 512;     cvt_bf16_k<<<(n + 255) / 256, 256, 0, stream>>>(w_obj, wobjB, n); }
    { int n = 36 * 512;    cvt_bf16_k<<<(n + 255) / 256, 256, 0, stream>>>(w_box, wboxB, n); }
    { int n = CCLS * 2048; cvt_bf16_k<<<(n + 255) / 256, 256, 0, stream>>>(w_cls, wclsB, n); }
    { int n = 512 * KRPN;  repack_wrpn_k<<<(n + 255) / 256, 256, 0, stream>>>(w_rpn, wrpnB, n); }

    // 2) backbone im2col (pure reshape)
    { int n = BATCH * NSP * KBB; im2col_bb_k<<<(n + 255) / 256, 256, 0, stream>>>(images, patch, n); }

    dim3 blk(32, 4, 1);
    // 3) backbone GEMM -> featsPad (bf16 padded NHWC), relu + bias fused
    gemm_wmma_k<0, 1, 1><<<dim3(4, 128, BATCH), blk, 0, stream>>>(
        wbbB, patch, b_bb, featsP, 2048, KBB,
        (long long)NSP * KBB, (long long)PADW * PADW * CFEAT);
    // 4) RPN 3x3 GEMM (K=9*2048 over padded NHWC) -> midT (bf16 NHWC), relu
    gemm_wmma_k<1, 2, 1><<<dim3(4, 32, BATCH), blk, 0, stream>>>(
        wrpnB, featsP, b_rpn, midT, 512, KRPN,
        (long long)PADW * PADW * CFEAT, (long long)NSP * CMID);
    // 5) obj 1x1 -> Cobj f32 (padded 16 rows)
    gemm_wmma_k<0, 0, 0><<<dim3(4, 1, BATCH), blk, 0, stream>>>(
        wobjB, midT, b_obj, Cobj, 16, 512,
        (long long)NSP * CMID, (long long)16 * NSP);
    // 6) box 1x1 -> Cbox f32 (padded 48 rows)
    gemm_wmma_k<0, 0, 0><<<dim3(4, 3, BATCH), blk, 0, stream>>>(
        wboxB, midT, b_box, Cbox, 48, 512,
        (long long)NSP * CMID, (long long)48 * NSP);
    // 7) cls 1x1 over featsPad -> Ccls f32 (padded 192 rows)
    gemm_wmma_k<2, 0, 0><<<dim3(4, 12, BATCH), blk, 0, stream>>>(
        wclsB, featsP, b_cls, Ccls, CCLS_P, 2048,
        (long long)PADW * PADW * CFEAT, (long long)CCLS_P * NSP);

    // 8) anchors + decode + clip + minsize
    { int n = BATCH * ANCH_TOT; decode_k<<<(n + 255) / 256, 256, 0, stream>>>(Cobj, Cbox, boxes, scoresA, n); }
    // 9) RPN NMS -> roi boxes
    rpn_nms_k<<<BATCH, 1024, 0, stream>>>(scoresA, boxes, roiB);
    // 10) PSROI pooling
    { int n = NROIS * NUM_CLASSES; psroi_k<<<(n + 255) / 256, 256, 0, stream>>>(Ccls, roiB, psS); }
    // 11) softmax
    softmax_k<<<(NROIS + 255) / 256, 256, 0, stream>>>(psS, probs);
    // 12) per-class NMS
    class_nms_k<<<20, 256, 0, stream>>>(probs, roiB, clsS, clsB);
    // 13) final top-100 -> d_out: boxes[400] | scores[100] | labels[100]
    final_topk_k<<<1, 256, 0, stream>>>(clsS, clsB, (float*)d_out);
}
